// GLASSConv_8143257994041
// MI455X (gfx1250) — matrix operand order repaired and checked by reading, not compiled, not run on hardware
//
#include <hip/hip_runtime.h>
#include <hip/hip_bf16.h>
#include <stdint.h>

// ---------------------------------------------------------------------------
// GLASSConv forward for MI455X (gfx1250, wave32, WMMA f16 -> f32)
// GEMMs: A fragments hoisted to registers, B tiles staged in LDS per block.
// ---------------------------------------------------------------------------
#define NNODE   50000
#define NEDGE   800000
#define DIN     256
#define DOUT    256
#define KCAT    512
#define EAdim   16
#define EHdim   32
#define ZMIX    0.8f
#define EPSN    1e-5f

typedef _Float16 v16h __attribute__((ext_vector_type(16)));
typedef _Float16 v8h  __attribute__((ext_vector_type(8)));
typedef float    v8f  __attribute__((ext_vector_type(8)));

static __device__ __forceinline__ v16h cat8(v8h lo, v8h hi) {
    return __builtin_shufflevector(lo, hi, 0,1,2,3,4,5,6,7,8,9,10,11,12,13,14,15);
}

// --------------------------- prep kernels ----------------------------------
// W [fin, fout] f32 row-major  ->  WT [fout, fin] f16 (K contiguous per row)
__global__ void k_transpose_f16(const float* __restrict__ W,
                                _Float16* __restrict__ WT,
                                int fin, int fout) {
    int idx = blockIdx.x * blockDim.x + threadIdx.x;
    if (idx >= fin * fout) return;
    int k = idx / fout;
    int n = idx - k * fout;
    WT[(size_t)n * fin + k] = (_Float16)W[idx];
}

__global__ void k_f32_to_f16(const float* __restrict__ src,
                             _Float16* __restrict__ dst, int total) {
    int i = blockIdx.x * blockDim.x + threadIdx.x;
    if (i < total) dst[i] = (_Float16)src[i];
}

// --------------------------- edge MLP --------------------------------------
__global__ __launch_bounds__(256)
void k_edge_mlp(const float* __restrict__ ea,      // [E,16]
                const float* __restrict__ Wm1,     // [16,32]
                const float* __restrict__ bm1,     // [32]
                const float* __restrict__ Wm2,     // [32,1]
                const float* __restrict__ bm2,     // [1]
                const int*   __restrict__ ei,      // [2,E]
                float* __restrict__ ew,            // [E]
                float* __restrict__ deg,           // [N] (zeroed)
                int nedge) {
    __shared__ float sW1[EAdim * EHdim];
    __shared__ float sb1[EHdim];
    __shared__ float sW2[EHdim];
    __shared__ float sb2;
    int t = threadIdx.x;
    for (int i = t; i < EAdim * EHdim; i += 256) sW1[i] = Wm1[i];
    if (t < EHdim) { sb1[t] = bm1[t]; sW2[t] = Wm2[t]; }
    if (t == 0) sb2 = bm2[0];
    __syncthreads();

    int e = blockIdx.x * 256 + t;
    if (e >= nedge) return;

    float a[EAdim];
#pragma unroll
    for (int i = 0; i < EAdim; ++i) a[i] = ea[(size_t)e * EAdim + i];

    float s = sb2;
#pragma unroll
    for (int j = 0; j < EHdim; ++j) {
        float h = sb1[j];
#pragma unroll
        for (int i = 0; i < EAdim; ++i) h += a[i] * sW1[i * EHdim + j];
        h = fmaxf(h, 0.0f);
        s += h * sW2[j];
    }
    // numerically stable softplus
    float sp = fmaxf(s, 0.0f) + log1pf(__expf(-fabsf(s)));
    ew[e] = sp;
    atomicAdd(&deg[ei[e]], sp);   // row = ei[0][e]
}

__global__ void k_deg_fix(float* __restrict__ deg, int n) {
    int i = blockIdx.x * blockDim.x + threadIdx.x;
    if (i < n) { float d = deg[i]; deg[i] = (d < 0.5f) ? d + 1.0f : d; }
}

// ------------------- transform GEMM (WMMA, LDS-staged B) -------------------
// xm = mix(mask, relu(x@Wt1+b1), relu(x@Wt0+b0))  stored f16 [N,256]
__global__ __launch_bounds__(256)
void k_transform_gemm(const _Float16* __restrict__ x16,  // [N,256]
                      const _Float16* __restrict__ W1T,  // [256,256] (fout,fin)
                      const _Float16* __restrict__ W0T,
                      const float* __restrict__ b1,
                      const float* __restrict__ b0,
                      const uint8_t* __restrict__ mask,  // [N]
                      _Float16* __restrict__ xm16,       // [N,256]
                      int nrows) {
    __shared__ _Float16 sB1[16 * DIN];   // one 16-col weight tile (8 KB)
    __shared__ _Float16 sB0[16 * DIN];

    const int lane = threadIdx.x & 31;
    const int wave = threadIdx.x >> 5;
    const int half = lane >> 4;       // K-half selector
    const int l16  = lane & 15;
    const int rowBase = blockIdx.x * 128 + wave * 16;

    int arow = rowBase + l16;
    if (arow >= nrows) arow = nrows - 1;          // clamp: keep EXEC all-ones
    const _Float16* aRow = x16 + (size_t)arow * DIN + half * 8;

    // Hoist all A fragments for this row tile (K = 256 -> 8 fragments)
    v16h aFrag[8];
#pragma unroll
    for (int kk = 0; kk < 8; ++kk) {
        const int kb = kk * 32;
        aFrag[kk] = cat8(*(const v8h*)(aRow + kb), *(const v8h*)(aRow + kb + 16));
    }

    for (int nt = 0; nt < DOUT; nt += 16) {
        // Cooperative stage of the contiguous 16x256 weight tiles into LDS
        {
            const _Float16* g1 = W1T + (size_t)nt * DIN;
            const _Float16* g0 = W0T + (size_t)nt * DIN;
#pragma unroll
            for (int i = 0; i < 2; ++i) {
                int idx = (threadIdx.x + i * 256) * 8;   // 4096 f16 per tile
                *(v8h*)&sB1[idx] = *(const v8h*)(g1 + idx);
                *(v8h*)&sB0[idx] = *(const v8h*)(g0 + idx);
            }
        }
        __syncthreads();

        v8f c1 = {}; v8f c0 = {};
        const _Float16* l1 = &sB1[l16 * DIN + half * 8];
        const _Float16* l0 = &sB0[l16 * DIN + half * 8];
#pragma unroll
        for (int kk = 0; kk < 8; ++kk) {
            const int kb = kk * 32;
            v16h w1 = cat8(*(const v8h*)(l1 + kb), *(const v8h*)(l1 + kb + 16));
            v16h w0 = cat8(*(const v8h*)(l0 + kb), *(const v8h*)(l0 + kb + 16));
            c1 = __builtin_amdgcn_wmma_f32_16x16x32_f16(false, aFrag[kk], false, w1,
                                                        (short)0, c1, false, false);
            c0 = __builtin_amdgcn_wmma_f32_16x16x32_f16(false, aFrag[kk], false, w0,
                                                        (short)0, c0, false, false);
        }

        const int n = nt + l16;
        const float bias1 = b1[n], bias0 = b0[n];
#pragma unroll
        for (int v = 0; v < 8; ++v) {
            int m = rowBase + half * 8 + v;       // D layout: lane half picks M block
            if (m < nrows) {
                float x1 = fmaxf(c1[v] + bias1, 0.0f);
                float x0 = fmaxf(c0[v] + bias0, 0.0f);
                float r  = mask[m] ? (ZMIX * x1 + (1.0f - ZMIX) * x0)
                                   : (ZMIX * x0 + (1.0f - ZMIX) * x1);
                xm16[(size_t)m * DOUT + n] = (_Float16)r;
            }
        }
        __syncthreads();   // protect LDS tile before next overwrite
    }
}

// ------------------- sparse aggregation (wave per edge) --------------------
__global__ __launch_bounds__(256)
void k_aggregate(const int* __restrict__ ei,
                 const float* __restrict__ ew,
                 const float* __restrict__ deg,
                 const _Float16* __restrict__ xm16,
                 float* __restrict__ agg,          // [N,256] zeroed
                 int nedge) {
    int e = blockIdx.x * 8 + (threadIdx.x >> 5);
    if (e >= nedge) return;
    const int lane = threadIdx.x & 31;
    const int r = ei[e];
    const int c = ei[nedge + e];
    const float w = ew[e] / deg[r];
    v8h v = *(const v8h*)(xm16 + (size_t)c * DOUT + lane * 8);
    float* dst = agg + (size_t)r * DOUT + lane * 8;
#pragma unroll
    for (int i = 0; i < 8; ++i) atomicAdd(dst + i, w * (float)v[i]);
}

// ------------------- GraphNorm: column stats + affine ----------------------
__global__ __launch_bounds__(256)
void k_colstats(const float* __restrict__ agg,
                float* __restrict__ colsum, float* __restrict__ colsq,
                int nrows, int rowsPer) {
    const int c  = threadIdx.x;                   // 256 columns
    const int r0 = blockIdx.x * rowsPer;
    int r1 = r0 + rowsPer; if (r1 > nrows) r1 = nrows;
    float s = 0.0f, s2 = 0.0f;
    for (int r = r0; r < r1; ++r) {
        float v = agg[(size_t)r * DOUT + c];
        s += v; s2 += v * v;
    }
    atomicAdd(&colsum[c], s);
    atomicAdd(&colsq[c], s2);
}

__global__ void k_norm_coef(const float* __restrict__ colsum,
                            const float* __restrict__ colsq,
                            const float* __restrict__ gw,
                            const float* __restrict__ gb,
                            const float* __restrict__ ga,
                            float* __restrict__ Ac, float* __restrict__ Bc) {
    int c = threadIdx.x;
    const float invN = 1.0f / (float)NNODE;
    float m   = colsum[c] * invN;
    float ex2 = colsq[c] * invN;
    float a   = ga[c];
    // var of (agg - a*m): E[agg^2] - 2*a*m*E[agg] + a^2*m^2
    float var = ex2 - 2.0f * a * m * m + a * a * m * m;
    float A   = gw[c] * rsqrtf(var + EPSN);
    Ac[c] = A;
    Bc[c] = gb[c] - A * a * m;
}

__global__ void k_norm_apply(const float* __restrict__ agg,
                             const float* __restrict__ Ac,
                             const float* __restrict__ Bc,
                             _Float16* __restrict__ out16, int total) {
    int i = blockIdx.x * blockDim.x + threadIdx.x;
    if (i >= total) return;
    int c = i & (DOUT - 1);
    out16[i] = (_Float16)(Ac[c] * agg[i] + Bc[c]);
}

// ------------------- output GEMM (K=512 concat, LDS-staged B) --------------
__global__ __launch_bounds__(256)
void k_out_gemm(const _Float16* __restrict__ an16,  // norm(agg) [N,256]
                const _Float16* __restrict__ x16,   // x f16     [N,256]
                const _Float16* __restrict__ Wc1T,  // [256,512] (fout,fin)
                const _Float16* __restrict__ Wc0T,
                const float* __restrict__ b1,
                const float* __restrict__ b0,
                const uint8_t* __restrict__ mask,
                float* __restrict__ out,            // [N,256] f32
                int nrows) {
    __shared__ _Float16 sB1[16 * KCAT];   // one 16-col weight tile (16 KB)
    __shared__ _Float16 sB0[16 * KCAT];

    const int lane = threadIdx.x & 31;
    const int wave = threadIdx.x >> 5;
    const int half = lane >> 4;
    const int l16  = lane & 15;
    const int rowBase = blockIdx.x * 128 + wave * 16;

    int arow = rowBase + l16;
    if (arow >= nrows) arow = nrows - 1;
    const size_t aOff = (size_t)arow * DIN + half * 8;

    // Hoist all A fragments (virtual concat: K 0..255 from an16, 256..511 from x16)
    v16h aFrag[16];
#pragma unroll
    for (int kk = 0; kk < 16; ++kk) {
        const int kb = kk * 32;
        const _Float16* ap = (kb < 256) ? (an16 + aOff + kb)
                                        : (x16 + aOff + (kb - 256));
        aFrag[kk] = cat8(*(const v8h*)(ap), *(const v8h*)(ap + 16));
    }

    for (int nt = 0; nt < DOUT; nt += 16) {
        {
            const _Float16* g1 = Wc1T + (size_t)nt * KCAT;
            const _Float16* g0 = Wc0T + (size_t)nt * KCAT;
#pragma unroll
            for (int i = 0; i < 4; ++i) {
                int idx = (threadIdx.x + i * 256) * 8;   // 8192 f16 per tile
                *(v8h*)&sB1[idx] = *(const v8h*)(g1 + idx);
                *(v8h*)&sB0[idx] = *(const v8h*)(g0 + idx);
            }
        }
        __syncthreads();

        v8f c1 = {}; v8f c0 = {};
        const _Float16* l1 = &sB1[l16 * KCAT + half * 8];
        const _Float16* l0 = &sB0[l16 * KCAT + half * 8];
#pragma unroll
        for (int kk = 0; kk < 16; ++kk) {
            const int kb = kk * 32;
            v16h w1 = cat8(*(const v8h*)(l1 + kb), *(const v8h*)(l1 + kb + 16));
            v16h w0 = cat8(*(const v8h*)(l0 + kb), *(const v8h*)(l0 + kb + 16));
            c1 = __builtin_amdgcn_wmma_f32_16x16x32_f16(false, aFrag[kk], false, w1,
                                                        (short)0, c1, false, false);
            c0 = __builtin_amdgcn_wmma_f32_16x16x32_f16(false, aFrag[kk], false, w0,
                                                        (short)0, c0, false, false);
        }

        const int n = nt + l16;
        const float bias1 = b1[n], bias0 = b0[n];
#pragma unroll
        for (int v = 0; v < 8; ++v) {
            int m = rowBase + half * 8 + v;
            if (m < nrows) {
                float y1 = c1[v] + bias1;
                float y0 = c0[v] + bias0;
                float r  = mask[m] ? (ZMIX * y1 + (1.0f - ZMIX) * y0)
                                   : (ZMIX * y0 + (1.0f - ZMIX) * y1);
                out[(size_t)m * DOUT + n] = r;
            }
        }
        __syncthreads();
    }
}

// ---------------------------------------------------------------------------
extern "C" void kernel_launch(void* const* d_in, const int* in_sizes, int n_in,
                              void* d_out, int out_size, void* d_ws, size_t ws_size,
                              hipStream_t stream) {
    // Inputs (setup_inputs order)
    const float*   x     = (const float*)d_in[0];
    const int*     ei    = (const int*)d_in[1];     // [2,E]
    /* d_in[2] edge_weight unused by reference */
    const uint8_t* mask  = (const uint8_t*)d_in[3]; // bool [N,1]
    const float*   eattr = (const float*)d_in[4];
    const float*   Wt0   = (const float*)d_in[5];
    const float*   bt0   = (const float*)d_in[6];
    const float*   Wt1   = (const float*)d_in[7];
    const float*   bt1   = (const float*)d_in[8];
    const float*   Wc0   = (const float*)d_in[9];
    const float*   bc0   = (const float*)d_in[10];
    const float*   Wc1   = (const float*)d_in[11];
    const float*   bc1   = (const float*)d_in[12];
    const float*   gn_w  = (const float*)d_in[13];
    const float*   gn_b  = (const float*)d_in[14];
    const float*   gn_a  = (const float*)d_in[15];
    const float*   Wm1   = (const float*)d_in[16];
    const float*   bm1   = (const float*)d_in[17];
    const float*   Wm2   = (const float*)d_in[18];
    const float*   bm2   = (const float*)d_in[19];

    // Workspace carve-up (256B aligned)
    uint8_t* ws = (uint8_t*)d_ws;
    size_t o = 0;
    auto take = [&](size_t bytes) { uint8_t* p = ws + o; o += (bytes + 255) & ~(size_t)255; return p; };
    _Float16* x16    = (_Float16*)take((size_t)NNODE * DIN  * 2);
    _Float16* xm16   = (_Float16*)take((size_t)NNODE * DOUT * 2);
    _Float16* an16   = (_Float16*)take((size_t)NNODE * DOUT * 2);
    float*    agg    = (float*)   take((size_t)NNODE * DOUT * 4);
    _Float16* W1T    = (_Float16*)take((size_t)DIN * DOUT * 2);
    _Float16* W0T    = (_Float16*)take((size_t)DIN * DOUT * 2);
    _Float16* Wc1T   = (_Float16*)take((size_t)KCAT * DOUT * 2);
    _Float16* Wc0T   = (_Float16*)take((size_t)KCAT * DOUT * 2);
    float*    ew     = (float*)   take((size_t)NEDGE * 4);
    float*    deg    = (float*)   take((size_t)NNODE * 4);
    float*    colsum = (float*)   take(DOUT * 4);
    float*    colsq  = (float*)   take(DOUT * 4);
    float*    Ac     = (float*)   take(DOUT * 4);
    float*    Bc     = (float*)   take(DOUT * 4);

    // Zero accumulators (graph-capturable memset nodes)
    hipMemsetAsync(agg,    0, (size_t)NNODE * DOUT * 4, stream);
    hipMemsetAsync(deg,    0, (size_t)NNODE * 4, stream);
    hipMemsetAsync(colsum, 0, DOUT * 4, stream);
    hipMemsetAsync(colsq,  0, DOUT * 4, stream);

    // 1) weight transpose/convert + x convert
    k_transpose_f16<<<(DIN * DOUT + 255) / 256, 256, 0, stream>>>(Wt1, W1T, DIN, DOUT);
    k_transpose_f16<<<(DIN * DOUT + 255) / 256, 256, 0, stream>>>(Wt0, W0T, DIN, DOUT);
    k_transpose_f16<<<(KCAT * DOUT + 255) / 256, 256, 0, stream>>>(Wc1, Wc1T, KCAT, DOUT);
    k_transpose_f16<<<(KCAT * DOUT + 255) / 256, 256, 0, stream>>>(Wc0, Wc0T, KCAT, DOUT);
    k_f32_to_f16<<<((size_t)NNODE * DIN + 255) / 256, 256, 0, stream>>>(x, x16, NNODE * DIN);

    // 2) edge MLP -> ew, degree accumulation; degree fix
    k_edge_mlp<<<(NEDGE + 255) / 256, 256, 0, stream>>>(eattr, Wm1, bm1, Wm2, bm2,
                                                        ei, ew, deg, NEDGE);
    k_deg_fix<<<(NNODE + 255) / 256, 256, 0, stream>>>(deg, NNODE);

    // 3) dual transform GEMM (WMMA) + masked mix -> xm16
    k_transform_gemm<<<(NNODE + 127) / 128, 256, 0, stream>>>(x16, W1T, W0T, bt1, bt0,
                                                              mask, xm16, NNODE);

    // 4) sparse aggregation (wave per edge)
    k_aggregate<<<(NEDGE + 7) / 8, 256, 0, stream>>>(ei, ew, deg, xm16, agg, NEDGE);

    // 5) GraphNorm: stats -> coefficients -> normalized f16 activations
    const int rowsPer = (NNODE + 255) / 256;
    k_colstats<<<256, 256, 0, stream>>>(agg, colsum, colsq, NNODE, rowsPer);
    k_norm_coef<<<1, 256, 0, stream>>>(colsum, colsq, gn_w, gn_b, gn_a, Ac, Bc);
    k_norm_apply<<<((size_t)NNODE * DOUT + 255) / 256, 256, 0, stream>>>(
        agg, Ac, Bc, an16, NNODE * DOUT);

    // 6) output GEMM (K=512 concat) + masked mix -> d_out (f32)
    k_out_gemm<<<(NNODE + 127) / 128, 256, 0, stream>>>(an16, x16, Wc1T, Wc0T,
                                                        bc1, bc0, mask,
                                                        (float*)d_out, NNODE);
}
